// PosteriorBernoulliSTBP_29265907155562
// MI455X (gfx1250) — matrix operand order repaired and checked by reading, not compiled, not run on hardware
//
#include <hip/hip_runtime.h>

// ---------------- problem constants (match reference setup_inputs) ----------
#define B_   512
#define C_   128
#define T_   16
#define K_   20
#define C2_  256    // 2*C
#define C4_  512    // 4*C
#define CK_  2560   // K*C

#define VTH_   0.2f
#define TAU_   0.25f
#define ALPHA_ 2.0f
#define EPS_   1e-5f

typedef __attribute__((ext_vector_type(16))) __bf16 v16bf;
typedef __attribute__((ext_vector_type(8)))  float  v8f;

__device__ __forceinline__ unsigned short f2bf(float f) {
    unsigned int u = __float_as_uint(f);
    u += 0x7FFFu + ((u >> 16) & 1u);   // round-to-nearest-even
    return (unsigned short)(u >> 16);
}

// CDNA5 async global->LDS copy (ASYNCcnt); 16B per lane.
__device__ __forceinline__ void async_ld_b128(unsigned lds_addr,
                                              const unsigned short* gptr) {
    asm volatile("global_load_async_to_lds_b128 %0, %1, off"
                 :: "v"(lds_addr), "v"(gptr)
                 : "memory");
}

// ---------------- fp32 -> bf16 weight conversion ----------------------------
__global__ void f32_to_bf16_kernel(const float* __restrict__ src,
                                   unsigned short* __restrict__ dst, int n) {
    int i = blockIdx.x * blockDim.x + threadIdx.x;
    if (i < n) dst[i] = f2bf(src[i]);
}

// ---------------- pack [x | z] prefix into bf16 activation matrix -----------
// A0 row r = b*L + l, cols 0..127 = x[b,c,l], cols 128..255 = z[b,c,l]
__global__ void pack_input_kernel(const float* __restrict__ x,
                                  const float* __restrict__ z,
                                  int L, unsigned short* __restrict__ A0) {
    int idx = blockIdx.x * blockDim.x + threadIdx.x;   // r*256 + col
    int col = idx & 255;
    int r   = idx >> 8;
    int b   = r / L;
    int l   = r - b * L;
    float v = (col < C_) ? x[((size_t)b * C_ + col) * T_ + l]
                         : z[((size_t)b * C_ + (col - C_)) * T_ + l];
    A0[idx] = f2bf(v);
}

// ---------------- bf16 WMMA GEMM:  Y[R,Cout] = A[R,Cin] * W[Cout,Cin]^T -----
// Block tile 128x128, K-step 32, 256 threads = 8 waves (2 in M x 4 in N),
// each wave computes 64x32 via 4x2 fragments of v_wmma_f32_16x16x32_bf16.
// Double-buffered LDS tiles filled by GLOBAL_LOAD_ASYNC_TO_LDS_B128; last two
// K-steps are peeled so the steady-state loop has no control flow.
#define BM_ 128
#define BN_ 128
#define BK_ 32
#define LDP_ 40   // padded LDS row stride (shorts): 80B, keeps 16B alignment

union Frag16 { uint4 u4[2]; v16bf v; };

__device__ __forceinline__ void load_frags(const unsigned short (*__restrict__ sAb)[LDP_],
                                           const unsigned short (*__restrict__ sBb)[LDP_],
                                           int wm, int wn, int lmod, int lhalf,
                                           Frag16 af[4], Frag16 bf[2]) {
#pragma unroll
    for (int j = 0; j < 2; ++j) {
        const int n = wn * 32 + j * 16 + lmod;
        bf[j].u4[0] = *(const uint4*)(&sBb[n][lhalf * 16]);
        bf[j].u4[1] = *(const uint4*)(&sBb[n][lhalf * 16 + 8]);
    }
#pragma unroll
    for (int i = 0; i < 4; ++i) {
        const int m = wm * 64 + i * 16 + lmod;
        af[i].u4[0] = *(const uint4*)(&sAb[m][lhalf * 16]);
        af[i].u4[1] = *(const uint4*)(&sAb[m][lhalf * 16 + 8]);
    }
}

__device__ __forceinline__ void do_wmma(const Frag16 af[4], const Frag16 bf[2],
                                        v8f acc[4][2]) {
#pragma unroll
    for (int i = 0; i < 4; ++i)
#pragma unroll
        for (int j = 0; j < 2; ++j)
            acc[i][j] = __builtin_amdgcn_wmma_f32_16x16x32_bf16(
                false, af[i].v, false, bf[j].v, (short)0, acc[i][j],
                false, false);
}

__global__ __launch_bounds__(256)
void gemm_bf16_wmma(const unsigned short* __restrict__ A,   // R x Cin
                    const unsigned short* __restrict__ W,   // Cout x Cin
                    float* __restrict__ Y,                  // R x Cout
                    int R, int Cin, int Cout) {
    __shared__ unsigned short sA[2][BM_][LDP_];
    __shared__ unsigned short sB[2][BN_][LDP_];

    const int bm   = blockIdx.x;
    const int bn   = blockIdx.y;
    const int tid  = threadIdx.x;
    const int wave = tid >> 5;
    const int lane = tid & 31;
    const int wm   = wave >> 2;        // 0..1 -> M offset wm*64
    const int wn   = wave & 3;         // 0..3 -> N offset wn*32
    const int lmod  = lane & 15;
    const int lhalf = lane >> 4;       // 0/1

    v8f acc[4][2];
#pragma unroll
    for (int i = 0; i < 4; ++i)
#pragma unroll
        for (int j = 0; j < 2; ++j)
#pragma unroll
            for (int r = 0; r < 8; ++r) acc[i][j][r] = 0.0f;

    // staging assignment: each thread copies 2x16B of A and 2x16B of W
    const int ldRow = tid >> 1;          // 0..127
    const int ldCol = (tid & 1) * 16;    // 0 or 16 (shorts)
    const unsigned short* aSrc = A + (size_t)(bm * BM_ + ldRow) * Cin + ldCol;
    const unsigned short* wSrc = W + (size_t)(bn * BN_ + ldRow) * Cin + ldCol;

    auto issue_tile = [&](int buf, int k0) {
        unsigned la = (unsigned)(uintptr_t)&sA[buf][ldRow][ldCol];
        unsigned lb = (unsigned)(uintptr_t)&sB[buf][ldRow][ldCol];
        async_ld_b128(la,      aSrc + k0);
        async_ld_b128(la + 16, aSrc + k0 + 8);
        async_ld_b128(lb,      wSrc + k0);
        async_ld_b128(lb + 16, wSrc + k0 + 8);
    };

    const int nTiles = Cin / BK_;        // >= 8 for all layers
    issue_tile(0, 0);
    issue_tile(1, BK_);

    Frag16 af[4], bf[2];

    // steady state: wait for current tile, prefetch tile it+2, compute
    for (int it = 0; it < nTiles - 2; ++it) {
        asm volatile("s_wait_asynccnt 0x4" ::: "memory");
        __syncthreads();
        const int buf = it & 1;
        load_frags(sA[buf], sB[buf], wm, wn, lmod, lhalf, af, bf);
        __syncthreads();                  // readers done before refill
        issue_tile(buf, (it + 2) * BK_);
        do_wmma(af, bf, acc);
    }
    // tile nTiles-2: nothing left to issue
    {
        asm volatile("s_wait_asynccnt 0x4" ::: "memory");
        __syncthreads();
        const int buf = (nTiles - 2) & 1;
        load_frags(sA[buf], sB[buf], wm, wn, lmod, lhalf, af, bf);
        do_wmma(af, bf, acc);
    }
    // last tile: drain the async queue completely; no refill, no second barrier
    {
        asm volatile("s_wait_asynccnt 0x0" ::: "memory");
        __syncthreads();
        const int buf = (nTiles - 1) & 1;
        load_frags(sA[buf], sB[buf], wm, wn, lmod, lhalf, af, bf);
        do_wmma(af, bf, acc);
    }

    // store: lane L -> N = lmod; VGPR r -> M = r + 8*lhalf
#pragma unroll
    for (int i = 0; i < 4; ++i) {
#pragma unroll
        for (int j = 0; j < 2; ++j) {
            const int gm = bm * BM_ + wm * 64 + i * 16 + lhalf * 8;
            const int gn = bn * BN_ + wn * 32 + j * 16 + lmod;
#pragma unroll
            for (int r = 0; r < 8; ++r)
                Y[(size_t)(gm + r) * Cout + gn] = acc[i][j][r];
        }
    }
}

// ---------------- per-channel batch stats (sum, sumsq) ----------------------
__global__ void bn_stats_kernel(const float* __restrict__ Y, int R, int Cout,
                                float* __restrict__ sum, float* __restrict__ ssq) {
    const int c  = blockIdx.x * 256 + threadIdx.x;   // Cout multiple of 256
    const int r0 = blockIdx.y * 256;
    int rend = r0 + 256; if (rend > R) rend = R;
    float s = 0.f, q = 0.f;
    for (int r = r0; r < rend; ++r) {
        float v = Y[(size_t)r * Cout + c];
        s += v; q += v * v;
    }
    atomicAdd(&sum[c], s);
    atomicAdd(&ssq[c], q);
}

// ---------------- BN normalize + LIF scan, emit bf16 spikes -----------------
__global__ void bn_lif_bf16_kernel(const float* __restrict__ Y, int L, int Cout,
                                   const float* __restrict__ sum,
                                   const float* __restrict__ ssq,
                                   const float* __restrict__ gamma,
                                   const float* __restrict__ beta,
                                   unsigned short* __restrict__ Aout) {
    const int idx = blockIdx.x * blockDim.x + threadIdx.x;  // b*Cout + o
    const int b = idx / Cout;
    const int o = idx - b * Cout;
    const float N  = (float)(B_ * L);
    const float m  = sum[o] / N;
    const float vr = ssq[o] / N - m * m;
    const float sc = ALPHA_ * VTH_ * gamma[o] * rsqrtf(vr + EPS_);
    const float sh = beta[o] - m * sc;
    float u = 0.f, sp = 0.f;
    for (int l = 0; l < L; ++l) {
        const size_t off = (size_t)(b * L + l) * Cout + o;
        const float y = Y[off] * sc + sh;
        u  = TAU_ * u * (1.f - sp) + y;
        sp = (u > VTH_) ? 1.f : 0.f;
        Aout[off] = f2bf(sp);
    }
}

// layer3 AR variant: keep only last-timestep spike as f32 -> qlast[b*CK + o]
__global__ void bn_lif3_ar_kernel(const float* __restrict__ Y, int L,
                                  const float* __restrict__ sum,
                                  const float* __restrict__ ssq,
                                  const float* __restrict__ gamma,
                                  const float* __restrict__ beta,
                                  float* __restrict__ qlast) {
    const int idx = blockIdx.x * blockDim.x + threadIdx.x;  // b*CK + o
    const int b = idx / CK_;
    const int o = idx - b * CK_;
    const float N  = (float)(B_ * L);
    const float m  = sum[o] / N;
    const float vr = ssq[o] / N - m * m;
    const float sc = ALPHA_ * VTH_ * gamma[o] * rsqrtf(vr + EPS_);
    const float sh = beta[o] - m * sc;
    float u = 0.f, sp = 0.f;
    for (int l = 0; l < L; ++l) {
        const float y = Y[(size_t)(b * L + l) * CK_ + o] * sc + sh;
        u  = TAU_ * u * (1.f - sp) + y;
        sp = (u > VTH_) ? 1.f : 0.f;
    }
    qlast[idx] = sp;
}

// layer3 final variant: write q_z[b, c, k, t] (= qz[(b*CK + o)*T + l], o=c*K+k)
__global__ void bn_lif3_final_kernel(const float* __restrict__ Y,
                                     const float* __restrict__ sum,
                                     const float* __restrict__ ssq,
                                     const float* __restrict__ gamma,
                                     const float* __restrict__ beta,
                                     float* __restrict__ qz) {
    const int idx = blockIdx.x * blockDim.x + threadIdx.x;  // b*CK + o
    const int b = idx / CK_;
    const int o = idx - b * CK_;
    const float N  = (float)(B_ * T_);
    const float m  = sum[o] / N;
    const float vr = ssq[o] / N - m * m;
    const float sc = ALPHA_ * VTH_ * gamma[o] * rsqrtf(vr + EPS_);
    const float sh = beta[o] - m * sc;
    float u = 0.f, sp = 0.f;
    for (int l = 0; l < T_; ++l) {
        const float y = Y[(size_t)(b * T_ + l) * CK_ + o] * sc + sh;
        u  = TAU_ * u * (1.f - sp) + y;
        sp = (u > VTH_) ? 1.f : 0.f;
        qz[(size_t)idx * T_ + l] = sp;
    }
}

// ---------------- sampling gathers ------------------------------------------
__global__ void sample_ar_kernel(const float* __restrict__ qlast,
                                 const int* __restrict__ rand_idx,
                                 int t, float* __restrict__ z) {
    const int r = blockIdx.x * blockDim.x + threadIdx.x;  // b*C + c
    const int k = rand_idx[t * (B_ * C_) + r];
    const int b = r / C_;
    const int c = r - b * C_;
    z[(size_t)r * T_ + (t + 1)] = qlast[(size_t)b * CK_ + c * K_ + k];
}

__global__ void sample_final_kernel(const float* __restrict__ qz,
                                    const int* __restrict__ rand_idx,
                                    float* __restrict__ out_sampled) {
    const int idx = blockIdx.x * blockDim.x + threadIdx.x;  // (b*C+c)*T + t
    const int t = idx & (T_ - 1);
    const int r = idx >> 4;
    const int k = rand_idx[t * (B_ * C_) + r];
    out_sampled[idx] = qz[((size_t)r * K_ + k) * T_ + t];
}

// ---------------- host orchestration ----------------------------------------
extern "C" void kernel_launch(void* const* d_in, const int* in_sizes, int n_in,
                              void* d_out, int out_size, void* d_ws, size_t ws_size,
                              hipStream_t stream) {
    (void)in_sizes; (void)n_in; (void)out_size; (void)ws_size;
    const float* x        = (const float*)d_in[0];
    const int*   rand_idx = (const int*)  d_in[1];
    const float* W1 = (const float*)d_in[2];
    const float* g1 = (const float*)d_in[4];
    const float* be1= (const float*)d_in[5];
    const float* W2 = (const float*)d_in[6];
    const float* g2 = (const float*)d_in[8];
    const float* be2= (const float*)d_in[9];
    const float* W3 = (const float*)d_in[10];
    const float* g3 = (const float*)d_in[12];
    const float* be3= (const float*)d_in[13];

    float* out_sampled = (float*)d_out;                      // (B,C,T)
    float* qz          = out_sampled + (size_t)B_ * C_ * T_; // (B,C,K,T)

    // workspace carve-out
    char* p = (char*)d_ws;
    auto carve = [&](size_t bytes) {
        void* r = (void*)p;
        p += (bytes + 255) & ~(size_t)255;
        return r;
    };
    unsigned short* Wb1 = (unsigned short*)carve((size_t)C2_ * C2_ * 2);
    unsigned short* Wb2 = (unsigned short*)carve((size_t)C4_ * C2_ * 2);
    unsigned short* Wb3 = (unsigned short*)carve((size_t)CK_ * C4_ * 2);
    unsigned short* A0  = (unsigned short*)carve((size_t)B_ * T_ * C2_ * 2);
    unsigned short* A1  = (unsigned short*)carve((size_t)B_ * T_ * C2_ * 2);
    unsigned short* A2  = (unsigned short*)carve((size_t)B_ * T_ * C4_ * 2);
    float* Ybuf  = (float*)carve((size_t)B_ * T_ * CK_ * 4);
    float* sumb  = (float*)carve((size_t)CK_ * 4);
    float* ssqb  = (float*)carve((size_t)CK_ * 4);
    float* zbuf  = (float*)carve((size_t)B_ * C_ * T_ * 4);
    float* qlast = (float*)carve((size_t)B_ * CK_ * 4);

    // weights -> bf16, z -> 0
    {
        int n1 = C2_ * C2_, n2 = C4_ * C2_, n3 = CK_ * C4_;
        f32_to_bf16_kernel<<<(n1 + 255) / 256, 256, 0, stream>>>(W1, Wb1, n1);
        f32_to_bf16_kernel<<<(n2 + 255) / 256, 256, 0, stream>>>(W2, Wb2, n2);
        f32_to_bf16_kernel<<<(n3 + 255) / 256, 256, 0, stream>>>(W3, Wb3, n3);
        hipMemsetAsync(zbuf, 0, (size_t)B_ * C_ * T_ * 4, stream);
    }

    auto run_stack = [&](int L, bool final_pass) {
        const int R = B_ * L;
        pack_input_kernel<<<(size_t)R * C2_ / 256, 256, 0, stream>>>(x, zbuf, L, A0);

        // ---- layer 1: (R x 256) x (256 x 256)^T
        gemm_bf16_wmma<<<dim3(R / BM_, C2_ / BN_), 256, 0, stream>>>(A0, Wb1, Ybuf, R, C2_, C2_);
        hipMemsetAsync(sumb, 0, C2_ * 4, stream);
        hipMemsetAsync(ssqb, 0, C2_ * 4, stream);
        bn_stats_kernel<<<dim3(C2_ / 256, R / 256), 256, 0, stream>>>(Ybuf, R, C2_, sumb, ssqb);
        bn_lif_bf16_kernel<<<(size_t)B_ * C2_ / 256, 256, 0, stream>>>(Ybuf, L, C2_, sumb, ssqb, g1, be1, A1);

        // ---- layer 2: (R x 256) x (512 x 256)^T
        gemm_bf16_wmma<<<dim3(R / BM_, C4_ / BN_), 256, 0, stream>>>(A1, Wb2, Ybuf, R, C2_, C4_);
        hipMemsetAsync(sumb, 0, C4_ * 4, stream);
        hipMemsetAsync(ssqb, 0, C4_ * 4, stream);
        bn_stats_kernel<<<dim3(C4_ / 256, R / 256), 256, 0, stream>>>(Ybuf, R, C4_, sumb, ssqb);
        bn_lif_bf16_kernel<<<(size_t)B_ * C4_ / 256, 256, 0, stream>>>(Ybuf, L, C4_, sumb, ssqb, g2, be2, A2);

        // ---- layer 3: (R x 512) x (2560 x 512)^T
        gemm_bf16_wmma<<<dim3(R / BM_, CK_ / BN_), 256, 0, stream>>>(A2, Wb3, Ybuf, R, C4_, CK_);
        hipMemsetAsync(sumb, 0, CK_ * 4, stream);
        hipMemsetAsync(ssqb, 0, CK_ * 4, stream);
        bn_stats_kernel<<<dim3(CK_ / 256, R / 256), 256, 0, stream>>>(Ybuf, R, CK_, sumb, ssqb);
        if (final_pass) {
            bn_lif3_final_kernel<<<(size_t)B_ * CK_ / 256, 256, 0, stream>>>(Ybuf, sumb, ssqb, g3, be3, qz);
        } else {
            bn_lif3_ar_kernel<<<(size_t)B_ * CK_ / 256, 256, 0, stream>>>(Ybuf, L, sumb, ssqb, g3, be3, qlast);
        }
    };

    // autoregressive sampling loop: t = 0..T-2
    for (int t = 0; t < T_ - 1; ++t) {
        run_stack(t + 1, /*final_pass=*/false);
        sample_ar_kernel<<<(B_ * C_) / 256, 256, 0, stream>>>(qlast, rand_idx, t, zbuf);
    }

    // full-length pass -> q_z, then gather sampled_z
    run_stack(T_, /*final_pass=*/true);
    sample_final_kernel<<<(size_t)B_ * C_ * T_ / 256, 256, 0, stream>>>(qz, rand_idx, out_sampled);
}
